// Covariance_Metric_49168785604781
// MI455X (gfx1250) — matrix operand order repaired and checked by reading, not compiled, not run on hardware
//
#include <hip/hip_runtime.h>
#include <hip/hip_bf16.h>

// ---------------------------------------------------------------------------
// Covariance-metric few-shot scorer for MI455X (gfx1250, wave32, WMMA).
// sim[q,w,i] = q_i^T Cov_w q_i (~173 GFLOP) -> bf16 WMMA 16x16x32, f32 accum.
// A-stationary in VGPRs (16 fragments/wave); B (packed cov fragments, 2.6 MB,
// L2-resident) is streamed via GLOBAL_LOAD_ASYNC_TO_LDS_B128 into a
// double-buffered 32 KB LDS stage shared by 4 waves (4 row-tiles) per block:
// one s_wait_asynccnt per 16 KB stage instead of s_wait_loadcnt 0 per WMMA,
// and 4x less L2 traffic. Epilogue fuses diag contraction, LeakyReLU(0.2)
// and the Conv1d(441) reduction into atomics on the (150,5) output.
// ---------------------------------------------------------------------------

typedef __bf16 v16bf __attribute__((ext_vector_type(16)));
typedef float  v8f   __attribute__((ext_vector_type(8)));
typedef unsigned short u16x8  __attribute__((ext_vector_type(8)));
typedef unsigned short u16x16 __attribute__((ext_vector_type(16)));

#define QN   150
#define WAYS 5
#define CH   512
#define HW   441
#define ROWP 448      // HW padded to 28 row-tiles of 16
#define NSUP 2205     // 5 shots * 441
#define NSUPP 2240    // padded to multiple of 32
#define NSTAGE (WAYS * 32)        // 160 (way, dtile) fragment sets
#define STAGE_BYTES 16384         // 16 ksteps * 32 lanes * 32 B

__device__ __forceinline__ unsigned short f2bf(float f) {
  unsigned int u = __builtin_bit_cast(unsigned int, f);
  u += 0x7FFFu + ((u >> 16) & 1u);      // round-to-nearest-even
  return (unsigned short)(u >> 16);
}
__device__ __forceinline__ float bf2f(unsigned short h) {
  return __builtin_bit_cast(float, ((unsigned int)h) << 16);
}
__device__ __forceinline__ v16bf make_frag(const unsigned short* p0,
                                           const unsigned short* p1) {
  u16x8 a = *(const u16x8*)p0;
  u16x8 b = *(const u16x8*)p1;
  u16x16 t;
#pragma unroll
  for (int j = 0; j < 8; ++j) { t[j] = a[j]; t[j + 8] = b[j]; }
  return __builtin_bit_cast(v16bf, t);
}

// --- CDNA5 async-to-LDS machinery ------------------------------------------
__device__ __forceinline__ unsigned lds_off(const void* p) {
  return (unsigned)(unsigned long long)
      (__attribute__((address_space(3))) const void*)p;
}
__device__ __forceinline__ void async_copy_b128(unsigned lds_byte,
                                                unsigned long long gaddr) {
  // VGLOBAL async op: VDST = LDS byte offset, VADDR = 64-bit global address.
  asm volatile("global_load_async_to_lds_b128 %0, %1, off"
               :: "v"(lds_byte), "v"(gaddr) : "memory");
}
__device__ __forceinline__ void wait_async0() {
#if __has_builtin(__builtin_amdgcn_s_wait_asynccnt)
  __builtin_amdgcn_s_wait_asynccnt(0);
#else
  asm volatile("s_wait_asynccnt 0x0" ::: "memory");
#endif
}

// ---- seed output with the conv bias (atomics accumulate on top) -----------
__global__ void k_init_out(float* out, const float* conv_b) {
  int i = blockIdx.x * blockDim.x + threadIdx.x;
  if (i < QN * WAYS) out[i] = conv_b[0];
}

// ---- per-(q,c) mean over HW: one wave each --------------------------------
__global__ void k_mean1(const float* x1, float* mean1) {
  int wid  = (blockIdx.x * blockDim.x + threadIdx.x) >> 5;
  int lane = threadIdx.x & 31;
  if (wid >= QN * CH) return;
  const float* p = x1 + (size_t)wid * HW;
  float s = 0.f;
  for (int i = lane; i < HW; i += 32) s += p[i];
#pragma unroll
  for (int m = 16; m; m >>= 1) s += __shfl_xor(s, m, 32);
  if (lane == 0) mean1[wid] = s * (1.0f / HW);
}

// ---- per-(way,c) mean over shot*HW: one wave each -------------------------
__global__ void k_mean2(const float* x2, float* mean2) {
  int wid  = (blockIdx.x * blockDim.x + threadIdx.x) >> 5;
  int lane = threadIdx.x & 31;
  if (wid >= WAYS * CH) return;
  int w = wid / CH, c = wid % CH;
  float s = 0.f;
  for (int sh = 0; sh < 5; ++sh) {
    const float* p = x2 + ((size_t)(w * 5 + sh) * CH + c) * HW;
    for (int i = lane; i < HW; i += 32) s += p[i];
  }
#pragma unroll
  for (int m = 16; m; m >>= 1) s += __shfl_xor(s, m, 32);
  if (lane == 0) mean2[wid] = s * (1.0f / NSUP);
}

// ---- centered bf16 query matrix, transposed (Q, 448, 512), LDS-tiled ------
__global__ void k_buildA(const float* x1, const float* mean1,
                         unsigned short* Abf) {
  __shared__ float tile[32][33];
  int q = blockIdx.z, i0 = blockIdx.x * 32, c0 = blockIdx.y * 32;
  int tx = threadIdx.x, ty = threadIdx.y;        // 32 x 8
#pragma unroll
  for (int j = 0; j < 4; ++j) {
    int c = c0 + ty + j * 8, i = i0 + tx;
    float v = 0.f;
    if (i < HW) v = x1[((size_t)q * CH + c) * HW + i] - mean1[q * CH + c];
    tile[ty + j * 8][tx] = v;
  }
  __syncthreads();
#pragma unroll
  for (int j = 0; j < 4; ++j) {
    int i = i0 + ty + j * 8, c = c0 + tx;
    Abf[((size_t)q * ROWP + i) * CH + c] = f2bf(tile[tx][ty + j * 8]);
  }
}

// ---- centered bf16 support matrix (way, 2240, 512), LDS-tiled -------------
__global__ void k_buildS(const float* x2, const float* mean2,
                         unsigned short* Sbf) {
  __shared__ float tile[32][33];
  int w = blockIdx.z, n0 = blockIdx.x * 32, c0 = blockIdx.y * 32;
  int tx = threadIdx.x, ty = threadIdx.y;
#pragma unroll
  for (int j = 0; j < 4; ++j) {
    int c = c0 + ty + j * 8, n = n0 + tx;
    float v = 0.f;
    if (n < NSUP) {
      int sh = n / HW, i = n % HW;
      v = x2[((size_t)(w * 5 + sh) * CH + c) * HW + i] - mean2[w * CH + c];
    }
    tile[ty + j * 8][tx] = v;
  }
  __syncthreads();
#pragma unroll
  for (int j = 0; j < 4; ++j) {
    int n = n0 + ty + j * 8, c = c0 + tx;
    Sbf[((size_t)w * NSUPP + n) * CH + c] = f2bf(tile[tx][ty + j * 8]);
  }
}

// ---- cov[w] = S^T S / (n-1), 16x16 output tile per block, LDS-staged ------
__global__ void k_cov(const unsigned short* Sbf, float* cov) {
  __shared__ float sc[32][17], sd[32][17];
  int w = blockIdx.z, d0 = blockIdx.x * 16, c0 = blockIdx.y * 16;
  int tx = threadIdx.x, ty = threadIdx.y;        // 16 x 16
  int tid = ty * 16 + tx;
  const unsigned short* S = Sbf + (size_t)w * NSUPP * CH;
  float acc = 0.f;
  for (int n0 = 0; n0 < NSUPP; n0 += 32) {
#pragma unroll
    for (int j = 0; j < 2; ++j) {
      int idx = tid + j * 256;
      int nn = idx >> 4, cc = idx & 15;
      sc[nn][cc] = bf2f(S[(size_t)(n0 + nn) * CH + c0 + cc]);
      sd[nn][cc] = bf2f(S[(size_t)(n0 + nn) * CH + d0 + cc]);
    }
    __syncthreads();
#pragma unroll
    for (int j = 0; j < 32; ++j) acc += sc[j][ty] * sd[j][tx];
    __syncthreads();
  }
  cov[((size_t)w * CH + c0 + ty) * CH + d0 + tx] = acc * (1.0f / (NSUP - 1));
}

// ---- pack cov into per-lane WMMA B fragments (32x16 bf16 each) ------------
// layout: [w][dtile(32)][kstep(16)][lane(32)] x 16 bf16 (32B per lane)
__global__ void k_packB(const float* cov, unsigned short* covPack) {
  int t = blockIdx.x * blockDim.x + threadIdx.x;  // 5*32*16*32 = 81920
  int lane = t & 31;
  int ks = (t >> 5) & 15;
  int dt = (t >> 9) & 31;
  int w  = t >> 14;
  if (w >= WAYS) return;
  int col = dt * 16 + (lane & 15);
  int kb  = ks * 32 + ((lane < 16) ? 0 : 16);
  const float* C = cov + (size_t)w * CH * CH;
  unsigned int* dst = (unsigned int*)covPack + (size_t)t * 8;
#pragma unroll
  for (int j = 0; j < 8; ++j) {
    unsigned short lo = f2bf(C[(size_t)(kb + 2 * j) * CH + col]);
    unsigned short hi = f2bf(C[(size_t)(kb + 2 * j + 1) * CH + col]);
    dst[j] = (unsigned int)lo | ((unsigned int)hi << 16);
  }
}

// ---- main: 4 waves/block (4 row tiles), async double-buffered B in LDS ----
// grid (7, 150), block 128. Per wave: 160 stages * 16 WMMA; B fragments are
// async-copied to LDS one stage ahead and shared by all 4 waves.
__global__ void __launch_bounds__(128)
k_main(const unsigned short* __restrict__ Abf,
       const unsigned short* __restrict__ covPack,
       const float* __restrict__ conv_w, float* __restrict__ out) {
  __shared__ unsigned short bstage[2][STAGE_BYTES / 2];  // 2 x 16 KB

  int q    = blockIdx.y;
  int tid  = threadIdx.x;
  int lane = tid & 31;
  int wave = tid >> 5;
  int tile = blockIdx.x * 4 + wave;
  int i0   = tile * 16;
  const unsigned short* Aq = Abf + (size_t)q * ROWP * CH;
  unsigned long long covAddr = (unsigned long long)(size_t)covPack;
  unsigned ldsBase = lds_off(&bstage[0][0]);

  // Resident A fragments: 16-bit A 16x32, contiguous k order (B packed the same way):
  // lanes 0-15: row=lane,    K = ks*32 + {0..15}
  // lanes16-31: row=lane-16, K = ks*32 + {16..31}
  v16bf afrag[16];
  int arow = i0 + (lane & 15);
  int koff = (lane < 16) ? 0 : 16;
#pragma unroll
  for (int ks = 0; ks < 16; ++ks) {
    const unsigned short* p = Aq + (size_t)arow * CH + ks * 32 + koff;
    afrag[ks] = make_frag(p, p + 8);
  }

  float simP[WAYS][8];
#pragma unroll
  for (int w = 0; w < WAYS; ++w)
#pragma unroll
    for (int r = 0; r < 8; ++r) simP[w][r] = 0.f;

  int erow0 = i0 + ((lane < 16) ? 0 : 8);  // C/D layout row base for this lane

  // prologue: stage 0 (w=0, dt=0) into buffer 0
#pragma unroll
  for (int j = 0; j < 8; ++j)
    async_copy_b128(ldsBase + j * 2048 + tid * 16,
                    covAddr + j * 2048 + tid * 16);

  for (int dt = 0; dt < 32; ++dt) {
    float ae[8];
#pragma unroll
    for (int w = 0; w < WAYS; ++w) {          // w fully unrolled: simP static
      int s = dt * WAYS + w;
      // current stage's async copies had all of stage s-1 to complete
      wait_async0();
      __syncthreads();   // everyone done reading buffer (s+1)&1 from stage s-1

      if (s + 1 < NSTAGE) {                   // issue next stage into other buf
        int ndt = (s + 1) / WAYS, nw = (s + 1) % WAYS;  // folds: w is constant
        unsigned long long g =
            covAddr + (unsigned long long)(nw * 32 + ndt) * STAGE_BYTES;
        unsigned l = ldsBase + (((s + 1) & 1) ? (unsigned)STAGE_BYTES : 0u);
#pragma unroll
        for (int j = 0; j < 8; ++j)
          async_copy_b128(l + j * 2048 + tid * 16, g + j * 2048 + tid * 16);
      }

      if (w == 0) {  // A values matching the C/D accumulator layout
        int col = dt * 16 + (lane & 15);
#pragma unroll
        for (int r = 0; r < 8; ++r)
          ae[r] = bf2f(Aq[(size_t)(erow0 + r) * CH + col]);
      }

      // consume stage s from buffer s&1, 1-step LDS rotation feeding WMMA
      const unsigned short* fb = &bstage[0][0] + (size_t)(s & 1) * (STAGE_BYTES / 2)
                                 + lane * 16;
      v8f acc = {};
      v16bf bcur = make_frag(fb, fb + 8);
#pragma unroll
      for (int ks = 0; ks < 16; ++ks) {
        v16bf bnext = bcur;
        if (ks < 15) {
          const unsigned short* p = fb + (ks + 1) * 512;
          bnext = make_frag(p, p + 8);
        }
        acc = __builtin_amdgcn_wmma_f32_16x16x32_bf16(
            false, afrag[ks], false, bcur, (short)0, acc, false, false);
        bcur = bnext;
      }
#pragma unroll
      for (int r = 0; r < 8; ++r) simP[w][r] += acc[r] * ae[r];
    }
  }

  // conv weights for this lane's 8 rows (zero for padded rows >= 441)
  float wc[8];
#pragma unroll
  for (int r = 0; r < 8; ++r) {
    int i = erow0 + r;
    wc[r] = (i < HW) ? conv_w[i] : 0.f;
  }

  // reduce sim over the 16 columns held per half-wave, then LeakyReLU + dot
#pragma unroll
  for (int w = 0; w < WAYS; ++w) {
    float contrib = 0.f;
#pragma unroll
    for (int r = 0; r < 8; ++r) {
      float s = simP[w][r];
#pragma unroll
      for (int m = 1; m < 16; m <<= 1) s += __shfl_xor(s, m, 32);
      float act = (s >= 0.f) ? s : 0.2f * s;
      contrib += act * wc[r];
    }
    if ((lane & 15) == 0) atomicAdd(&out[q * WAYS + w], contrib);
  }
}

// ---------------------------------------------------------------------------
extern "C" void kernel_launch(void* const* d_in, const int* in_sizes, int n_in,
                              void* d_out, int out_size, void* d_ws,
                              size_t ws_size, hipStream_t stream) {
  const float* x1     = (const float*)d_in[0];  // (150,512,21,21)
  const float* x2     = (const float*)d_in[1];  // (25,512,21,21)
  const float* conv_w = (const float*)d_in[2];  // (441,)
  const float* conv_b = (const float*)d_in[3];  // (1,)
  float* out = (float*)d_out;                   // (150,5)

  char* ws = (char*)d_ws;
  const size_t O_MEAN2 = 0;         //  5*512*4        = 10240
  const size_t O_MEAN1 = 10240;     //  150*512*4      = 307200
  const size_t O_COV   = 317440;    //  5*512*512*4    = 5242880
  const size_t O_PACK  = 5560320;   //  5*512*512*2    = 2621440
  const size_t O_SBF   = 8181760;   //  5*2240*512*2   = 11468800
  const size_t O_ABF   = 19650560;  //  150*448*512*2  = 68812800  (end 88463360)

  float*          mean2   = (float*)(ws + O_MEAN2);
  float*          mean1   = (float*)(ws + O_MEAN1);
  float*          cov     = (float*)(ws + O_COV);
  unsigned short* covPack = (unsigned short*)(ws + O_PACK);
  unsigned short* Sbf     = (unsigned short*)(ws + O_SBF);
  unsigned short* Abf     = (unsigned short*)(ws + O_ABF);

  k_init_out<<<3, 256, 0, stream>>>(out, conv_b);
  k_mean2<<<(WAYS * CH * 32) / 256, 256, 0, stream>>>(x2, mean2);
  k_mean1<<<(QN * CH * 32) / 256, 256, 0, stream>>>(x1, mean1);
  k_buildA<<<dim3(14, 16, QN), dim3(32, 8), 0, stream>>>(x1, mean1, Abf);
  k_buildS<<<dim3(70, 16, WAYS), dim3(32, 8), 0, stream>>>(x2, mean2, Sbf);
  k_cov<<<dim3(32, 32, WAYS), dim3(16, 16), 0, stream>>>(Sbf, cov);
  k_packB<<<(WAYS * 32 * 16 * 32) / 256, 256, 0, stream>>>(cov, covPack);
  k_main<<<dim3(7, QN), 128, 0, stream>>>(Abf, covPack, conv_w, out);
}
